// CrossAttention_50130858279651
// MI455X (gfx1250) — matrix operand order repaired and checked by reading, not compiled, not run on hardware
//
#include <hip/hip_runtime.h>

// ---------------------------------------------------------------------------
// CrossAttention for MI455X (gfx1250, wave32, WMMA)
//   B=8, S=4096, D=1024
//   - all GEMMs on v_wmma_f32_16x16x32_bf16 (f32 accumulate)
//   - projections write Q,K transposed ([b][d][s]) so the big scores GEMM
//     (K-dim = 4096) stages both operands with pure b128 copies ->
//     GLOBAL_LOAD_ASYNC_TO_LDS_B128 (ASYNCcnt) on this toolchain
//   - double-buffered LDS, one barrier per K-step
// ---------------------------------------------------------------------------

#define BDIM 256
#define LDSS 40            // LDS row stride in bf16 elems (32 data + 8 pad -> 80B)

static constexpr int Bn = 8;
static constexpr int Sn = 4096;
static constexpr int Dn = 1024;

typedef __attribute__((ext_vector_type(16))) __bf16        bf16x16;
typedef __attribute__((ext_vector_type(8)))  float         f32x8;
typedef __attribute__((ext_vector_type(4)))  float         f32x4;
typedef __attribute__((ext_vector_type(4)))  unsigned int  u32x4;
typedef __attribute__((ext_vector_type(4)))  int           i32x4;

union FragU { u32x4 q[2]; bf16x16 v; };

__device__ __forceinline__ unsigned short f2bf(float f) {
  unsigned u = __builtin_bit_cast(unsigned, f);
  u += 0x7FFFu + ((u >> 16) & 1u);          // round-to-nearest-even
  return (unsigned short)(u >> 16);
}
__device__ __forceinline__ unsigned pk2(float a, float b) {
  return (unsigned)f2bf(a) | ((unsigned)f2bf(b) << 16);
}

// ---- async copy global->LDS (CDNA5 ASYNCcnt path, guarded) ----------------
#if defined(__has_builtin)
#  if __has_builtin(__builtin_amdgcn_global_load_async_to_lds_b128)
#    define ASYNC_LDS 1
#  endif
#endif
#ifndef ASYNC_LDS
#  define ASYNC_LDS 0
#endif

__device__ __forceinline__ void cp_b128(const unsigned short* g, unsigned short* l) {
#if ASYNC_LDS
  __builtin_amdgcn_global_load_async_to_lds_b128(
      (__attribute__((address_space(1))) i32x4*)g,
      (__attribute__((address_space(3))) i32x4*)l, 0, 0);
#else
  *(u32x4*)l = *(const u32x4*)g;
#endif
}
__device__ __forceinline__ void wait_fill() {
#if ASYNC_LDS
#  if __has_builtin(__builtin_amdgcn_s_wait_asynccnt)
  __builtin_amdgcn_s_wait_asynccnt(0);
#  else
  asm volatile("s_wait_asynccnt 0" ::: "memory");
#  endif
#endif
}

// ---- LDS tile fills -------------------------------------------------------
// Row-major bf16 copy: 128 rows x 32 k (2 x b128 per thread).
__device__ __forceinline__ void fill_cp(unsigned short* lds, const unsigned short* __restrict__ src,
                                        int ld, int m0, int k0) {
  int t = threadIdx.x;
  int r = t >> 1;
  int h = (t & 1) << 4;
  const unsigned short* g = src + (size_t)(m0 + r) * ld + k0 + h;
  unsigned short* l = lds + r * LDSS + h;
  cp_b128(g, l);
  cp_b128(g + 8, l + 8);
}

// Row-major f32 source -> bf16 LDS.
__device__ __forceinline__ void fill_rm_f32(unsigned short* lds, const float* __restrict__ src,
                                            int ld, int m0, int k0) {
  int t = threadIdx.x;
  int r = t >> 1;
  int h = (t & 1) << 4;
  const float* p = src + (size_t)(m0 + r) * ld + k0 + h;
  f32x4 f0 = ((const f32x4*)p)[0];
  f32x4 f1 = ((const f32x4*)p)[1];
  f32x4 f2 = ((const f32x4*)p)[2];
  f32x4 f3 = ((const f32x4*)p)[3];
  u32x4 a, b;
  a.x = pk2(f0.x, f0.y); a.y = pk2(f0.z, f0.w);
  a.z = pk2(f1.x, f1.y); a.w = pk2(f1.z, f1.w);
  b.x = pk2(f2.x, f2.y); b.y = pk2(f2.z, f2.w);
  b.z = pk2(f3.x, f3.y); b.w = pk2(f3.z, f3.w);
  u32x4* d = (u32x4*)(lds + r * LDSS + h);
  d[0] = a; d[1] = b;
}

// Source row-major [k][n] f32, stored transposed [n][k] bf16.
__device__ __forceinline__ void fill_tr_f32(unsigned short* lds, const float* __restrict__ src,
                                            int ld, int k0, int n0) {
  int t = threadIdx.x;
  int k = t >> 3;
  int g = (t & 7) << 4;
  const float* p = src + (size_t)(k0 + k) * ld + n0 + g;
  float v[16];
#pragma unroll
  for (int i = 0; i < 4; i++) {
    f32x4 f = ((const f32x4*)p)[i];
    v[i * 4 + 0] = f.x; v[i * 4 + 1] = f.y; v[i * 4 + 2] = f.z; v[i * 4 + 3] = f.w;
  }
#pragma unroll
  for (int i = 0; i < 16; i++) lds[(g + i) * LDSS + k] = f2bf(v[i]);
}

// Source row-major [k][n] bf16, stored transposed [n][k] bf16.
__device__ __forceinline__ void fill_tr_bf16(unsigned short* lds, const unsigned short* __restrict__ src,
                                             int ld, int k0, int n0) {
  int t = threadIdx.x;
  int k = t >> 3;
  int g = (t & 7) << 4;
  const unsigned short* p = src + (size_t)(k0 + k) * ld + n0 + g;
  unsigned short v[16];
  *(u32x4*)&v[0] = ((const u32x4*)p)[0];
  *(u32x4*)&v[8] = ((const u32x4*)p)[1];
#pragma unroll
  for (int i = 0; i < 16; i++) lds[(g + i) * LDSS + k] = v[i];
}

// ---- WMMA fragment loads (CDNA5 ISA 7.12.2 layouts) -----------------------
__device__ __forceinline__ bf16x16 load_frag_a(const unsigned short* lds, int row0) {
  int lane = threadIdx.x & 31;
  int m = lane & 15;
  int kh = lane >> 4;
  const unsigned short* p = lds + (row0 + m) * LDSS + (kh << 3);
  FragU f;
  f.q[0] = *(const u32x4*)p;
  f.q[1] = *(const u32x4*)(p + 16);
  return f.v;
}
__device__ __forceinline__ bf16x16 load_frag_b(const unsigned short* lds, int n0) {
  int lane = threadIdx.x & 31;
  int n = lane & 15;
  int kh = lane >> 4;
  const unsigned short* p = lds + (n0 + n) * LDSS + (kh << 3);
  FragU f;
  f.q[0] = *(const u32x4*)p;
  f.q[1] = *(const u32x4*)(p + 16);
  return f.v;
}

__device__ __forceinline__ f32x8 wmma_bf16(bf16x16 a, bf16x16 b, f32x8 c) {
  return __builtin_amdgcn_wmma_f32_16x16x32_bf16(false, a, false, b, (short)0, c, false, false);
}

__device__ __forceinline__ void gemm_step(const unsigned short* A, const unsigned short* B,
                                          int wm, int wn, f32x8 (&acc)[4][2]) {
  bf16x16 af[4], bfr[2];
#pragma unroll
  for (int mi = 0; mi < 4; mi++) af[mi] = load_frag_a(A, wm + (mi << 4));
#pragma unroll
  for (int ni = 0; ni < 2; ni++) bfr[ni] = load_frag_b(B, wn + (ni << 4));
#pragma unroll
  for (int mi = 0; mi < 4; mi++)
#pragma unroll
    for (int ni = 0; ni < 2; ni++)
      acc[mi][ni] = wmma_bf16(af[mi], bfr[ni], acc[mi][ni]);
}

// ---- output stores --------------------------------------------------------
__device__ __forceinline__ void store_frag_f32(float* __restrict__ C, int ldc,
                                               int row0, int col0, f32x8 acc) {
  int lane = threadIdx.x & 31;
  int col = col0 + (lane & 15);
  int rb  = row0 + ((lane >> 4) << 3);
#pragma unroll
  for (int r = 0; r < 8; r++) C[(size_t)(rb + r) * ldc + col] = acc[r];
}
// Normal bf16 store [m][n].
__device__ __forceinline__ void store_frag_bf16(unsigned short* __restrict__ C, int ldc,
                                                int row0, int col0, f32x8 acc) {
  int lane = threadIdx.x & 31;
  int col = col0 + (lane & 15);
  int rb  = row0 + ((lane >> 4) << 3);
#pragma unroll
  for (int r = 0; r < 8; r++) C[(size_t)(rb + r) * ldc + col] = f2bf(acc[r]);
}
// Transposed bf16 store: global row m = b*Sn+s -> O[b][col][s]; 8 rows are
// contiguous in s -> one packed b128 store per fragment.
__device__ __forceinline__ void store_frag_bf16_T(unsigned short* __restrict__ O,
                                                  int row0, int col0, f32x8 acc) {
  int lane = threadIdx.x & 31;
  int col = col0 + (lane & 15);
  int rb  = row0 + ((lane >> 4) << 3);
  int b   = rb >> 12;                  // Sn = 4096
  int s   = rb & (Sn - 1);
  unsigned short* p = O + ((size_t)b * Dn + col) * Sn + s;
  u32x4 v;
  v.x = pk2(acc[0], acc[1]); v.y = pk2(acc[2], acc[3]);
  v.z = pk2(acc[4], acc[5]); v.w = pk2(acc[6], acc[7]);
  *(u32x4*)p = v;
}

#define GEMM_PROLOG()                                   \
  __shared__ unsigned short ldsA[2][128 * LDSS];        \
  __shared__ unsigned short ldsB[2][128 * LDSS];        \
  int wave = threadIdx.x >> 5;                          \
  int wm = (wave >> 2) << 6;                            \
  int wn = (wave & 3) << 5;                             \
  f32x8 acc[4][2];                                      \
  _Pragma("unroll") for (int mi = 0; mi < 4; mi++)      \
  _Pragma("unroll") for (int ni = 0; ni < 2; ni++)      \
  _Pragma("unroll") for (int e = 0; e < 8; e++) acc[mi][ni][e] = 0.0f;

// ---- kernels --------------------------------------------------------------
// Projection: O = X @ W.  X f32 [32768,1024], W f32 [1024,1024].
// TRANSPOSED=true writes O as [b][n][s] bf16 (for Q,K), else [m][n] bf16 (V).
template <bool TRANSPOSED>
__global__ __launch_bounds__(BDIM) void proj_kernel(const float* __restrict__ X,
                                                    const float* __restrict__ W,
                                                    unsigned short* __restrict__ O) {
  int m0 = blockIdx.y * 128;
  int n0 = blockIdx.x * 128;
  GEMM_PROLOG();
  fill_rm_f32(ldsA[0], X, Dn, m0, 0);
  fill_tr_f32(ldsB[0], W, Dn, 0, n0);
  const int NT = Dn / 32;
  for (int kt = 0; kt < NT; kt++) {
    int cur = kt & 1;
    __syncthreads();
    if (kt + 1 < NT) {
      __builtin_prefetch(X + (size_t)(m0 + (threadIdx.x >> 1)) * Dn + (kt + 2) * 32, 0, 1);
      fill_rm_f32(ldsA[cur ^ 1], X, Dn, m0, (kt + 1) * 32);
      fill_tr_f32(ldsB[cur ^ 1], W, Dn, (kt + 1) * 32, n0);
    }
    gemm_step(ldsA[cur], ldsB[cur], wm, wn, acc);
  }
#pragma unroll
  for (int mi = 0; mi < 4; mi++)
#pragma unroll
    for (int ni = 0; ni < 2; ni++) {
      if (TRANSPOSED)
        store_frag_bf16_T(O, m0 + wm + (mi << 4), n0 + wn + (ni << 4), acc[mi][ni]);
      else
        store_frag_bf16(O, Dn, m0 + wm + (mi << 4), n0 + wn + (ni << 4), acc[mi][ni]);
    }
}

// Scores: SC[b,d,e] = sum_s Qt[b,d,s] * Kt[b,e,s].  Pure-copy fills (async path).
__global__ __launch_bounds__(BDIM) void scores_kernel(const unsigned short* __restrict__ Qt,
                                                      const unsigned short* __restrict__ Kt,
                                                      float* __restrict__ SC) {
  int b = blockIdx.z;
  const unsigned short* Qb = Qt + (size_t)b * Dn * Sn;   // [d][s]
  const unsigned short* Kb = Kt + (size_t)b * Dn * Sn;   // [e][s]
  float* Sb = SC + (size_t)b * Dn * Dn;
  int m0 = blockIdx.y * 128;   // d
  int n0 = blockIdx.x * 128;   // e
  GEMM_PROLOG();
  fill_cp(ldsA[0], Qb, Sn, m0, 0);
  fill_cp(ldsB[0], Kb, Sn, n0, 0);
  const int NT = Sn / 32;
  for (int kt = 0; kt < NT; kt++) {
    int cur = kt & 1;
    wait_fill();
    __syncthreads();
    if (kt + 1 < NT) {
      fill_cp(ldsA[cur ^ 1], Qb, Sn, m0, (kt + 1) * 32);
      fill_cp(ldsB[cur ^ 1], Kb, Sn, n0, (kt + 1) * 32);
    }
    gemm_step(ldsA[cur], ldsB[cur], wm, wn, acc);
  }
#pragma unroll
  for (int mi = 0; mi < 4; mi++)
#pragma unroll
    for (int ni = 0; ni < 2; ni++)
      store_frag_f32(Sb, Dn, m0 + wm + (mi << 4), n0 + wn + (ni << 4), acc[mi][ni]);
}

// Context: OUT[b,s,e] = sum_d V[b,s,d] * AW[b,d,e].  A-fill async copy, B-fill transpose.
__global__ __launch_bounds__(BDIM) void ctx_kernel(const unsigned short* __restrict__ V,
                                                   const unsigned short* __restrict__ AW,
                                                   float* __restrict__ OUT) {
  int b = blockIdx.z;
  const unsigned short* Vb = V + (size_t)b * Sn * Dn;
  const unsigned short* Ab = AW + (size_t)b * Dn * Dn;
  float* Ob = OUT + (size_t)b * Sn * Dn;
  int m0 = blockIdx.y * 128;   // s
  int n0 = blockIdx.x * 128;   // e
  GEMM_PROLOG();
  fill_cp(ldsA[0], Vb, Dn, m0, 0);
  fill_tr_bf16(ldsB[0], Ab, Dn, 0, n0);
  const int NT = Dn / 32;
  for (int kt = 0; kt < NT; kt++) {
    int cur = kt & 1;
    wait_fill();
    __syncthreads();
    if (kt + 1 < NT) {
      __builtin_prefetch(Ab + (size_t)((kt + 2) * 32 + (threadIdx.x >> 3)) * Dn + n0, 0, 1);
      fill_cp(ldsA[cur ^ 1], Vb, Dn, m0, (kt + 1) * 32);
      fill_tr_bf16(ldsB[cur ^ 1], Ab, Dn, (kt + 1) * 32, n0);
    }
    gemm_step(ldsA[cur], ldsB[cur], wm, wn, acc);
  }
#pragma unroll
  for (int mi = 0; mi < 4; mi++)
#pragma unroll
    for (int ni = 0; ni < 2; ni++)
      store_frag_f32(Ob, Dn, m0 + wm + (mi << 4), n0 + wn + (ni << 4), acc[mi][ni]);
}

// ---- softmax over last axis of [8,1024,1024], scale 1/sqrt(1024) ----------
__global__ __launch_bounds__(BDIM) void softmax_kernel(const float* __restrict__ SC,
                                                       unsigned short* __restrict__ AW) {
  int row = blockIdx.x;
  const float* src = SC + (size_t)row * Dn;
  unsigned short* dst = AW + (size_t)row * Dn;
  int t = threadIdx.x;
  __shared__ float red[BDIM];

  float v[4];
  float mx = -3.4e38f;
#pragma unroll
  for (int i = 0; i < 4; i++) {
    v[i] = src[t + i * BDIM] * 0.03125f;       // 1/sqrt(1024)
    mx = fmaxf(mx, v[i]);
  }
  red[t] = mx;
  __syncthreads();
  for (int s = BDIM / 2; s > 0; s >>= 1) {
    if (t < s) red[t] = fmaxf(red[t], red[t + s]);
    __syncthreads();
  }
  mx = red[0];
  __syncthreads();

  float sum = 0.0f;
#pragma unroll
  for (int i = 0; i < 4; i++) {
    v[i] = __expf(v[i] - mx);
    sum += v[i];
  }
  red[t] = sum;
  __syncthreads();
  for (int s = BDIM / 2; s > 0; s >>= 1) {
    if (t < s) red[t] += red[t + s];
    __syncthreads();
  }
  float inv = 1.0f / red[0];
#pragma unroll
  for (int i = 0; i < 4; i++) dst[t + i * BDIM] = f2bf(v[i] * inv);
}

// ---------------------------------------------------------------------------
extern "C" void kernel_launch(void* const* d_in, const int* in_sizes, int n_in,
                              void* d_out, int out_size, void* d_ws, size_t ws_size,
                              hipStream_t stream) {
  const float* x1 = (const float*)d_in[0];
  const float* x2 = (const float*)d_in[1];
  const float* Wq = (const float*)d_in[2];
  const float* Wk = (const float*)d_in[3];
  const float* Wv = (const float*)d_in[4];
  float* out = (float*)d_out;

  // Workspace (~240 MB): Qt,Kt [b][d][s] bf16, V [b][s][d] bf16, scores f32, attn bf16
  const size_t QKV = (size_t)Bn * Sn * Dn * 2;
  const size_t SCB = (size_t)Bn * Dn * Dn * 4;
  char* ws = (char*)d_ws;
  unsigned short* Qt = (unsigned short*)(ws);
  unsigned short* Kt = (unsigned short*)(ws + QKV);
  unsigned short* V  = (unsigned short*)(ws + 2 * QKV);
  float*          SC = (float*)(ws + 3 * QKV);
  unsigned short* AW = (unsigned short*)(ws + 3 * QKV + SCB);

  dim3 blk(BDIM);
  dim3 gproj(Dn / 128, (Bn * Sn) / 128);          // 8 x 256
  proj_kernel<true ><<<gproj, blk, 0, stream>>>(x1, Wq, Qt);
  proj_kernel<true ><<<gproj, blk, 0, stream>>>(x2, Wk, Kt);
  proj_kernel<false><<<gproj, blk, 0, stream>>>(x2, Wv, V);

  dim3 gsc(Dn / 128, Dn / 128, Bn);               // 8 x 8 x 8
  scores_kernel<<<gsc, blk, 0, stream>>>(Qt, Kt, SC);

  softmax_kernel<<<dim3(Bn * Dn), blk, 0, stream>>>(SC, AW);

  dim3 gcx(Dn / 128, Sn / 128, Bn);               // 8 x 32 x 8
  ctx_kernel<<<gcx, blk, 0, stream>>>(V, AW, out);
}
